// HINGE_56985626083396
// MI455X (gfx1250) — matrix operand order repaired and checked by reading, not compiled, not run on hardware
//
#include <hip/hip_runtime.h>
#include <hip/hip_bf16.h>

typedef __attribute__((ext_vector_type(16))) __bf16 v16bf;
typedef __attribute__((ext_vector_type(8)))  float  v8f;

#define B_TOT   32768
#define E_DIM   256
#define NFILT   512
#define CH      3072          // 6 * 512 channels
#define BN_EPS  1e-5f

#if defined(__has_builtin)
#  if __has_builtin(__builtin_amdgcn_sched_barrier)
#    define SCHED_FENCE() __builtin_amdgcn_sched_barrier(0)
#  endif
#endif
#ifndef SCHED_FENCE
#  define SCHED_FENCE()
#endif

// ---------- helpers ----------
__device__ __forceinline__ unsigned short f2bf(float f) {
  union { float f; unsigned u; } v; v.f = f;
  unsigned u = v.u;
  u += 0x7FFFu + ((u >> 16) & 1u);   // round-to-nearest-even bf16
  return (unsigned short)(u >> 16);
}
__device__ __forceinline__ unsigned pack2(float a, float b) {
  return (unsigned)f2bf(a) | ((unsigned)f2bf(b) << 16);
}

// ---------- weight conversion f32 -> bf16 ----------
__global__ void cvt_w_kernel(const float* __restrict__ w,
                             unsigned short* __restrict__ o, int n) {
  int i = blockIdx.x * 256 + threadIdx.x;
  if (i < n) o[i] = f2bf(w[i]);
}

__global__ void zero_kernel(float* __restrict__ p, int n) {
  int i = blockIdx.x * 256 + threadIdx.x;
  if (i < n) p[i] = 0.f;
}

// ---------- WMMA GEMM ----------
// Block: 256 threads = 8 waves. Block tile: M=32 x N=512 (full filter dim).
// Each wave owns a 32(M) x 64(N) strip -> 8 C tiles of 16x16 (64 acc VGPRs).
// Per K-step of 32: 2 A fragments (LDS) + 4 B fragments (global, software-
// pipelined one step ahead behind a sched_barrier) feed 8 WMMAs.
// A rows are gathered embedding rows (bf16, staged per 256-K segment in LDS).
// B is the bf16 conv weight matrix, filter-row-major == K-contiguous.
template <int SEGS, int KSLOT_BASE>
__global__ __launch_bounds__(256, 1)
void gemm_kernel(const int* __restrict__ xb,
                 const float* __restrict__ embR,
                 const float* __restrict__ embV,
                 const unsigned short* __restrict__ wbf,
                 float* __restrict__ allw) {
  constexpr int K = SEGS * 256;
  __shared__ __align__(32) unsigned short aSeg[32][256];   // 16 KB

  const int tid  = threadIdx.x;
  const int nw   = tid >> 5;          // 0..7 : 64-col strip of this wave
  const int lane = tid & 31;
  const int rowBase = blockIdx.x * 32;

  const int m     = lane & 15;          // A-frag row within 16-row group
  const int hk    = (lane >> 4) * 16;   // A-frag half-K select (contiguous K)
  const int kHalf = (lane >> 4) * 16;   // B-frag half-K select
  const int nLane = nw * 64 + (lane & 15);

  v8f acc[8] = {v8f{}, v8f{}, v8f{}, v8f{}, v8f{}, v8f{}, v8f{}, v8f{}};

  for (int s = 0; s < SEGS; ++s) {
    // ---- stage segment s of the A tile (32 rows x 256 bf16) ----
    {
      const int r  = tid >> 3;          // 0..31 (row in tile)
      const int e0 = (tid & 7) * 32;    // 32 floats per thread
      const int gr = rowBase + r;
      const float* src;
      if constexpr (SEGS == 2) {
        const int b = gr >> 1, j = gr & 1;
        // row (b,j) = [ fv_j | fr_0 ]
        src = (s == 0) ? embV + (size_t)xb[b * 12 + 2 * j + 1] * E_DIM
                       : embR + (size_t)xb[b * 12 + 0] * E_DIM;
      } else {
        const int b = gr >> 2, k = gr & 3;
        // row (b,k) = [ fv0 | fr0 | fv1 | kr_k | kv_k ]
        if      (s == 0) src = embV + (size_t)xb[b * 12 + 1] * E_DIM;
        else if (s == 1) src = embR + (size_t)xb[b * 12 + 0] * E_DIM;
        else if (s == 2) src = embV + (size_t)xb[b * 12 + 3] * E_DIM;
        else if (s == 3) src = embR + (size_t)xb[b * 12 + 4 + 2 * k] * E_DIM;
        else             src = embV + (size_t)xb[b * 12 + 5 + 2 * k] * E_DIM;
      }
      const float4* s4 = (const float4*)(src + e0);
      uint2* dst = (uint2*)&aSeg[r][e0];
#pragma unroll
      for (int i = 0; i < 8; ++i) {
        float4 f = s4[i];
        uint2 p;
        p.x = pack2(f.x, f.y);
        p.y = pack2(f.z, f.w);
        dst[i] = p;
      }
    }
    __syncthreads();

    // ---- prime B fragments for ks = 0 ----
    v16bf bcur[4];
#pragma unroll
    for (int nt = 0; nt < 4; ++nt)
      bcur[nt] = *(const v16bf*)(const void*)(
          wbf + (size_t)(nLane + nt * 16) * K + s * 256 + kHalf);

    // ---- 8 WMMA K-steps of 32, B prefetched one step ahead ----
#pragma unroll
    for (int ks = 0; ks < 8; ++ks) {
      // prefetch next step's B fragments (must not sink into the WMMA group)
      v16bf bnxt[4];
      if (ks < 7) {
        const int kGlob = s * 256 + (ks + 1) * 32 + kHalf;
#pragma unroll
        for (int nt = 0; nt < 4; ++nt)
          bnxt[nt] = *(const v16bf*)(const void*)(
              wbf + (size_t)(nLane + nt * 16) * K + kGlob);
      }

      // A fragments: contiguous K order matching B:
      //   lanes 0-15 : M=lane,    K[0:16) in v0-7
      //   lanes 16-31: M=lane-16, K[16:32) in v0-7
      union { uint4 u[2]; v16bf v; } a0, a1;
      a0.u[0] = *(const uint4*)(const void*)&aSeg[m][ks * 32 + hk];
      a0.u[1] = *(const uint4*)(const void*)&aSeg[m][ks * 32 + 8 + hk];
      a1.u[0] = *(const uint4*)(const void*)&aSeg[16 + m][ks * 32 + hk];
      a1.u[1] = *(const uint4*)(const void*)&aSeg[16 + m][ks * 32 + 8 + hk];

      SCHED_FENCE();   // keep prefetches above, WMMAs below

#pragma unroll
      for (int nt = 0; nt < 4; ++nt) {
        acc[nt]     = __builtin_amdgcn_wmma_f32_16x16x32_bf16(
            false, a0.v, false, bcur[nt], (short)0, acc[nt],     false, false);
        acc[4 + nt] = __builtin_amdgcn_wmma_f32_16x16x32_bf16(
            false, a1.v, false, bcur[nt], (short)0, acc[4 + nt], false, false);
      }

      SCHED_FENCE();   // keep next iteration's loads after this WMMA group

      if (ks < 7) {
#pragma unroll
        for (int nt = 0; nt < 4; ++nt) bcur[nt] = bnxt[nt];
      }
    }
    __syncthreads();
  }

  // ---- store C tiles: VGPR v holds (M = v + 8*(lane>=16), N = lane&15) ----
#pragma unroll
  for (int mt = 0; mt < 2; ++mt) {
#pragma unroll
    for (int nt = 0; nt < 4; ++nt) {
      const int n = nw * 64 + nt * 16 + (lane & 15);
#pragma unroll
      for (int v = 0; v < 8; ++v) {
        const int mrow = mt * 16 + v + ((lane >> 4) << 3);
        const int gr   = rowBase + mrow;
        int b, kslot;
        if constexpr (SEGS == 2) { b = gr >> 1; kslot = KSLOT_BASE + (gr & 1); }
        else                     { b = gr >> 2; kslot = KSLOT_BASE + (gr & 3); }
        allw[(size_t)b * CH + kslot * NFILT + n] = acc[mt * 4 + nt][v];
      }
    }
  }
}

// ---------- per-channel sum / sumsq over the batch ----------
__global__ void stats_kernel(const float* __restrict__ allw,
                             float* __restrict__ sum,
                             float* __restrict__ sumsq) {
  const int t = threadIdx.x;                       // 256 threads
  const size_t row0 = (size_t)blockIdx.x * 64;     // 512 blocks x 64 rows
  float s1[12] = {}, s2[12] = {};
  for (int r = 0; r < 64; ++r) {
    const float* p = allw + (row0 + r) * CH;
#pragma unroll
    for (int i = 0; i < 12; ++i) {
      float v = p[t + i * 256];
      s1[i] += v;
      s2[i] += v * v;
    }
  }
#pragma unroll
  for (int i = 0; i < 12; ++i) {
    atomicAdd(&sum[t + i * 256],   s1[i]);
    atomicAdd(&sumsq[t + i * 256], s2[i]);
  }
}

// ---------- fold mean/var + gamma/beta into scale/shift ----------
__global__ void finalize_stats_kernel(const float* __restrict__ sum,
                                      const float* __restrict__ sumsq,
                                      const float* __restrict__ g1,
                                      const float* __restrict__ b1,
                                      const float* __restrict__ g2,
                                      const float* __restrict__ b2,
                                      float* __restrict__ scale,
                                      float* __restrict__ shift) {
  const int c = blockIdx.x * 256 + threadIdx.x;
  if (c >= CH) return;
  const float inv_b = 1.f / (float)B_TOT;
  const float m   = sum[c] * inv_b;
  const float var = sumsq[c] * inv_b - m * m;
  const int f = c & (NFILT - 1);
  const float g  = (c < 2 * NFILT) ? g1[f] : g2[f];
  const float be = (c < 2 * NFILT) ? b1[f] : b2[f];
  const float sc = g * rsqrtf(var + BN_EPS);
  scale[c] = sc;
  shift[c] = be - m * sc;
}

// ---------- BN + relu + min over 6 slots + fc dot ----------
__global__ void final_kernel(const float* __restrict__ allw,
                             const float* __restrict__ scale,
                             const float* __restrict__ shift,
                             const float* __restrict__ fcw,
                             const float* __restrict__ fcb,
                             float* __restrict__ out) {
  const int lane = threadIdx.x & 31;
  const size_t b = (size_t)blockIdx.x * 8 + (threadIdx.x >> 5);
  const float* p = allw + b * CH;
  float acc = 0.f;
  for (int f = lane; f < NFILT; f += 32) {
    float mn = 3.4e38f;
#pragma unroll
    for (int ks = 0; ks < 6; ++ks) {
      const int c = ks * NFILT + f;
      float v = p[c] * scale[c] + shift[c];
      v = v > 0.f ? v : 0.f;        // relu
      mn = v < mn ? v : mn;         // min over slots
    }
    acc += mn * fcw[f];
  }
#pragma unroll
  for (int off = 16; off; off >>= 1) acc += __shfl_xor(acc, off, 32);
  if (lane == 0) out[b] = acc + fcb[0];
}

// ---------- host launch ----------
extern "C" void kernel_launch(void* const* d_in, const int* in_sizes, int n_in,
                              void* d_out, int out_size, void* d_ws, size_t ws_size,
                              hipStream_t stream) {
  const int*   xb     = (const int*)  d_in[0];   // (B, 12) int32
  // d_in[1] = arity (6, constant)
  const float* embR   = (const float*)d_in[2];   // (NR, 256)
  const float* embV   = (const float*)d_in[3];   // (NV, 256)
  const float* c1w    = (const float*)d_in[4];   // (512, 2, 256)
  // d_in[5] = conv1_b : cancels exactly under batch-norm (mean subtraction)
  const float* g1     = (const float*)d_in[6];
  const float* be1    = (const float*)d_in[7];
  const float* c2w    = (const float*)d_in[8];   // (512, 5, 256)
  // d_in[9] = conv2_b : cancels under batch-norm
  const float* g2     = (const float*)d_in[10];
  const float* be2    = (const float*)d_in[11];
  const float* fcw    = (const float*)d_in[12];  // (1, 512)
  const float* fcb    = (const float*)d_in[13];  // (1,)
  float* out = (float*)d_out;

  // workspace layout
  char* ws = (char*)d_ws;
  const size_t ALLW_OFF  = 0;                                 // B*CH f32 = 384 MiB
  const size_t W1_OFF    = (size_t)B_TOT * CH * 4;            // 512*512 bf16
  const size_t W2_OFF    = W1_OFF + (size_t)NFILT * 512 * 2;  // 512*1280 bf16
  const size_t SUM_OFF   = W2_OFF + (size_t)NFILT * 1280 * 2;
  const size_t SSQ_OFF   = SUM_OFF + CH * 4;
  const size_t SCALE_OFF = SSQ_OFF + CH * 4;
  const size_t SHIFT_OFF = SCALE_OFF + CH * 4;

  float*          allw  = (float*)(ws + ALLW_OFF);
  unsigned short* wbf1  = (unsigned short*)(ws + W1_OFF);
  unsigned short* wbf2  = (unsigned short*)(ws + W2_OFF);
  float*          sum   = (float*)(ws + SUM_OFF);
  float*          sumsq = (float*)(ws + SSQ_OFF);
  float*          scale = (float*)(ws + SCALE_OFF);
  float*          shift = (float*)(ws + SHIFT_OFF);

  // 1. weights -> bf16 (K-contiguous per filter row, matches B-fragment layout)
  cvt_w_kernel<<<(NFILT * 512 + 255) / 256, 256, 0, stream>>>(c1w, wbf1, NFILT * 512);
  cvt_w_kernel<<<(NFILT * 1280 + 255) / 256, 256, 0, stream>>>(c2w, wbf2, NFILT * 1280);

  // 2. zero the stats accumulators (every call: graph replays need this)
  zero_kernel<<<(2 * CH + 255) / 256, 256, 0, stream>>>(sum, 2 * CH);

  // 3. WMMA GEMMs: conv1 (M=2B, K=512) and conv2 (M=4B, K=1280), N=512
  gemm_kernel<2, 0><<<(2 * B_TOT) / 32, 256, 0, stream>>>(
      xb, embR, embV, wbf1, allw);
  gemm_kernel<5, 2><<<(4 * B_TOT) / 32, 256, 0, stream>>>(
      xb, embR, embV, wbf2, allw);

  // 4. batch statistics per channel
  stats_kernel<<<512, 256, 0, stream>>>(allw, sum, sumsq);

  // 5. fold into scale/shift
  finalize_stats_kernel<<<(CH + 255) / 256, 256, 0, stream>>>(
      sum, sumsq, g1, be1, g2, be2, scale, shift);

  // 6. BN + relu + min + fc
  final_kernel<<<B_TOT / 8, 256, 0, stream>>>(allw, scale, shift, fcw, fcb, out);

  (void)in_sizes; (void)n_in; (void)out_size; (void)ws_size;
}
